// TemporalInputAttention_54494545051891
// MI455X (gfx1250) — matrix operand order repaired and checked by reading, not compiled, not run on hardware
//
#include <hip/hip_runtime.h>
#include <hip/hip_bf16.h>

// ---------------- problem constants ----------------
#define H_    8
#define V_    25
#define C_IN  64
#define C_OUT 64
#define N_    32
#define T_    256
#define DIN   1600   // V*C_IN
#define DOUT  1600   // V*C_OUT
#define NT_   (N_*T_) // 8192

typedef __bf16 bf16_t;
typedef __attribute__((ext_vector_type(16))) __bf16 bf16x16;
typedef __attribute__((ext_vector_type(8)))  __bf16 bf16x8;
typedef __attribute__((ext_vector_type(8)))  float  f32x8;

// float -> bf16 round-to-nearest-even
__device__ __forceinline__ bf16_t f2bf(float f) {
  union { float f; unsigned u; } in; in.f = f;
  unsigned u = in.u;
  u += 0x7fffu + ((u >> 16) & 1u);
  unsigned short hs = (unsigned short)(u >> 16);
  union { unsigned short s; bf16_t b; } out; out.s = hs;
  return out.b;
}

// ---------------- WMMA fragment loaders (wave32, 16x16x32 bf16) ----------------
// A (16x32, MxK): lane m=lane&15 holds row m; hh=lane>>4 selects K-halves:
//   elems[0..7]  = K[k0 + hh*8 .. +7], elems[8..15] = K[k0+16+hh*8 .. +7]
__device__ __forceinline__ bf16x16 load_frag_a(const bf16_t* __restrict__ base,
                                               long row0, int ld, int k0, int lane) {
  int m  = lane & 15;
  int hh = lane >> 4;
  const bf16_t* p = base + (row0 + m) * (long)ld + k0 + hh * 8;
  bf16x8 lo = *(const bf16x8*)p;
  bf16x8 hi = *(const bf16x8*)(p + 16);
  return __builtin_shufflevector(lo, hi, 0,1,2,3,4,5,6,7,8,9,10,11,12,13,14,15);
}

// B (32x16, KxN): lane n=lane&15 holds column n; elems[0..7] = K[k0+hh*8 .. +7], elems[8..15] = K[k0+16+hh*8 .. +7].
// `base` stores B pre-transposed: base[col][k] contiguous along k with pitch ld.
__device__ __forceinline__ bf16x16 load_frag_b(const bf16_t* __restrict__ base,
                                               long col0, int ld, int k0, int lane) {
  int n  = lane & 15;
  int hh = lane >> 4;
  const bf16_t* p = base + (col0 + n) * (long)ld + k0 + hh * 8;
  bf16x8 lo = *(const bf16x8*)p;
  bf16x8 hi = *(const bf16x8*)(p + 16);
  return __builtin_shufflevector(lo, hi, 0,1,2,3,4,5,6,7,8,9,10,11,12,13,14,15);
}

__device__ __forceinline__ f32x8 wmma_bf16(bf16x16 a, bf16x16 b, f32x8 c) {
  // (neg_a, A, neg_b, B, c_mod, C, reuse_a, reuse_b)
  return __builtin_amdgcn_wmma_f32_16x16x32_bf16(false, a, false, b, (short)0, c,
                                                 false, false);
}

// Per-wave tile: 64 rows x 32 cols (4x2 tiles, 8 wmma per K-step).
// Double-buffered fragments: 64 (acc) + 96 (2x frag sets) + addressing ~= 175 VGPRs.
#define MI_ 4
#define NJ_ 2

template <int MI, int NJ>
__device__ __forceinline__ void load_frags(const bf16_t* __restrict__ Abase, long row0, int lda,
                                           const bf16_t* __restrict__ Bbase, long col0, int ldb,
                                           int k, int lane, bf16x16 (&av)[MI], bf16x16 (&bv)[NJ]) {
#pragma unroll
  for (int i = 0; i < MI; ++i) av[i] = load_frag_a(Abase, row0 + i * 16, lda, k, lane);
#pragma unroll
  for (int j = 0; j < NJ; ++j) bv[j] = load_frag_b(Bbase, col0 + j * 16, ldb, k, lane);
}

template <int MI, int NJ>
__device__ __forceinline__ void mma_tiles(f32x8 (&acc)[MI][NJ],
                                          const bf16x16 (&av)[MI], const bf16x16 (&bv)[NJ]) {
#pragma unroll
  for (int i = 0; i < MI; ++i)
#pragma unroll
    for (int j = 0; j < NJ; ++j)
      acc[i][j] = wmma_bf16(av[i], bv[j], acc[i][j]);
}

// Full K loop with ping-pong double buffering. Requires K/32 even and >= 2.
template <int MI, int NJ>
__device__ __forceinline__ void gemm_loop(const bf16_t* __restrict__ Abase, long row0, int lda,
                                          const bf16_t* __restrict__ Bbase, long col0, int ldb,
                                          int K, int lane, f32x8 (&acc)[MI][NJ]) {
  bf16x16 aP[MI], bP[NJ], aQ[MI], bQ[NJ];
  load_frags<MI, NJ>(Abase, row0, lda, Bbase, col0, ldb, 0, lane, aP, bP);
  int k = 32;
  for (; k + 32 < K; k += 64) {
    load_frags<MI, NJ>(Abase, row0, lda, Bbase, col0, ldb, k, lane, aQ, bQ);
    mma_tiles<MI, NJ>(acc, aP, bP);
    load_frags<MI, NJ>(Abase, row0, lda, Bbase, col0, ldb, k + 32, lane, aP, bP);
    mma_tiles<MI, NJ>(acc, aQ, bQ);
  }
  // epilogue: steps S-2 (in P) and S-1
  load_frags<MI, NJ>(Abase, row0, lda, Bbase, col0, ldb, K - 32, lane, aQ, bQ);
  mma_tiles<MI, NJ>(acc, aP, bP);
  mma_tiles<MI, NJ>(acc, aQ, bQ);
}

// ---------------- stage 1: pack [n,c,t,v] f32 -> [n*t, c*v] bf16 ----------------
__global__ void pack_nctv(const float* __restrict__ src, bf16_t* __restrict__ dst) {
  long idx = (long)blockIdx.x * blockDim.x + threadIdx.x; // over N*T*C*V, output-major
  int v = (int)(idx % V_);
  long t1 = idx / V_;
  int c = (int)(t1 % C_IN);
  long t2 = t1 / C_IN;
  int t = (int)(t2 % T_);
  int n = (int)(t2 / T_);
  float val = src[(((long)(n * C_IN + c)) * T_ + t) * V_ + v];
  dst[idx] = f2bf(val);
}

// ---------------- stage 2: W[h][f][o] f32 -> Wt[h][o][f] bf16 (LDS tiled) -------
__global__ void transpose_w(const float* __restrict__ W, bf16_t* __restrict__ Wt) {
  __shared__ float tile[32][33];
  int h  = blockIdx.z;
  int f0 = blockIdx.x * 32;
  int o0 = blockIdx.y * 32;
  const float* Wh  = W  + (long)h * DIN * DOUT;
  bf16_t*      Wth = Wt + (long)h * DIN * DOUT;
  int tx = threadIdx.x, ty = threadIdx.y; // 32 x 8
#pragma unroll
  for (int j = 0; j < 32; j += 8)
    tile[ty + j][tx] = Wh[(long)(f0 + ty + j) * DOUT + (o0 + tx)];
  __syncthreads();
#pragma unroll
  for (int j = 0; j < 32; j += 8)
    Wth[(long)(o0 + ty + j) * DIN + (f0 + tx)] = f2bf(tile[tx][ty + j]);
}

// ---------------- stage 3: per-head projection GEMM -----------------------------
// out = A[NT,DIN] x Wt[h][o][f]^T + bias[h][o]
// OUT_T==0: out bf16 [n,h,t,o] ; OUT_T==1: out bf16 [n,h,o,t]
template <int OUT_T>
__global__ __launch_bounds__(32)
void proj_gemm(const bf16_t* __restrict__ A,
               const bf16_t* __restrict__ Wt,
               const float*  __restrict__ bias,
               bf16_t* __restrict__ out) {
  int lane = threadIdx.x;
  long r0 = (long)blockIdx.x * (MI_ * 16);
  long c0 = (long)blockIdx.y * (NJ_ * 16);
  int  h  = blockIdx.z;
  const bf16_t* Bb = Wt + (long)h * DOUT * DIN;
  f32x8 acc[MI_][NJ_] = {};
  gemm_loop<MI_, NJ_>(A, r0, DIN, Bb, c0, DIN, DIN, lane, acc);

  int n  = lane & 15;
  int hh = lane >> 4;
#pragma unroll
  for (int i = 0; i < MI_; ++i)
#pragma unroll
    for (int j = 0; j < NJ_; ++j) {
      int gc = (int)c0 + j * 16 + n;
      float bs = bias[h * DOUT + gc];
#pragma unroll
      for (int r = 0; r < 8; ++r) {
        int grow = (int)r0 + i * 16 + hh * 8 + r;
        int nb = grow >> 8;          // / T_
        int t  = grow & (T_ - 1);
        float val = acc[i][j][r] + bs;
        long oidx;
        if (OUT_T == 0)
          oidx = ((long)(nb * H_ + h) * T_ + t) * DOUT + gc;
        else
          oidx = ((long)(nb * H_ + h) * DOUT + gc) * T_ + t;
        out[oidx] = f2bf(val);
      }
    }
}

// ---------------- stage 4: att[n,h,q,k] = scale * Q.K^T --------------------------
__global__ __launch_bounds__(32)
void attn_scores(const bf16_t* __restrict__ Q,
                 const bf16_t* __restrict__ K,
                 float* __restrict__ att) {
  int lane = threadIdx.x;
  long q0  = (long)blockIdx.x * (MI_ * 16);
  long k0c = (long)blockIdx.y * (NJ_ * 16);
  int  nh  = blockIdx.z;
  const bf16_t* Qb = Q + (long)nh * T_ * DOUT;
  const bf16_t* Kb = K + (long)nh * T_ * DOUT; // rows of K == columns of B
  f32x8 acc[MI_][NJ_] = {};
  gemm_loop<MI_, NJ_>(Qb, q0, DOUT, Kb, k0c, DOUT, DOUT, lane, acc);

  const float scale = 0.025f;  // 1/sqrt(1600)
  int n = lane & 15, hh = lane >> 4;
#pragma unroll
  for (int i = 0; i < MI_; ++i)
#pragma unroll
    for (int j = 0; j < NJ_; ++j)
#pragma unroll
      for (int r = 0; r < 8; ++r) {
        int q  = (int)q0 + i * 16 + hh * 8 + r;
        int kc = (int)k0c + j * 16 + n;
        att[((long)nh * T_ + q) * T_ + kc] = acc[i][j][r] * scale;
      }
}

// ---------------- stage 5: softmax over HEADS axis -------------------------------
__global__ void softmax_heads(const float* __restrict__ att, bf16_t* __restrict__ attb) {
  long idx = (long)blockIdx.x * blockDim.x + threadIdx.x; // over N*T*T
  int k = (int)(idx & (T_ - 1));
  long t1 = idx >> 8;
  int q = (int)(t1 & (T_ - 1));
  int n = (int)(t1 >> 8);
  long base = (((long)n * H_) * T_ + q) * T_ + k;
  const long hs = (long)T_ * T_;
  float v[H_];
  float mx = -1e30f;
#pragma unroll
  for (int h = 0; h < H_; ++h) { v[h] = att[base + h * hs]; mx = fmaxf(mx, v[h]); }
  float s = 0.f;
#pragma unroll
  for (int h = 0; h < H_; ++h) { v[h] = __expf(v[h] - mx); s += v[h]; }
  float inv = 1.0f / s;
#pragma unroll
  for (int h = 0; h < H_; ++h) attb[base + h * hs] = f2bf(v[h] * inv);
}

// ---------------- stage 6: y = att @ V, scatter to [n, c*H+h, t, v] f32 ----------
__global__ __launch_bounds__(32)
void attn_apply(const bf16_t* __restrict__ attb,
                const bf16_t* __restrict__ Vt,   // [n,h,o,t]
                float* __restrict__ out) {
  int lane = threadIdx.x;
  long q0 = (long)blockIdx.x * (MI_ * 16);
  long d0 = (long)blockIdx.y * (NJ_ * 16);
  int  nh = blockIdx.z;
  int  nb = nh >> 3, h = nh & 7;
  const bf16_t* Ab = attb + (long)nh * T_ * T_;
  const bf16_t* Bb = Vt   + (long)nh * DOUT * T_;
  f32x8 acc[MI_][NJ_] = {};
  gemm_loop<MI_, NJ_>(Ab, q0, T_, Bb, d0, T_, T_, lane, acc);

  int n = lane & 15, hh = lane >> 4;
#pragma unroll
  for (int i = 0; i < MI_; ++i)
#pragma unroll
    for (int j = 0; j < NJ_; ++j)
#pragma unroll
      for (int r = 0; r < 8; ++r) {
        int t = (int)q0 + i * 16 + hh * 8 + r;
        int d = (int)d0 + j * 16 + n;
        int vv = d >> 6;        // / C_OUT
        int c  = d & 63;        // % C_OUT
        long oidx = (((long)(nb * (C_OUT * H_) + c * H_ + h)) * T_ + t) * V_ + vv;
        out[oidx] = acc[i][j][r];
      }
}

// ---------------- launcher -------------------------------------------------------
extern "C" void kernel_launch(void* const* d_in, const int* in_sizes, int n_in,
                              void* d_out, int out_size, void* d_ws, size_t ws_size,
                              hipStream_t stream) {
  const float* x  = (const float*)d_in[0];
  const float* mm = (const float*)d_in[1];
  const float* Wk = (const float*)d_in[2];
  const float* bk = (const float*)d_in[3];
  const float* Wq = (const float*)d_in[4];
  const float* bq = (const float*)d_in[5];
  const float* Wv = (const float*)d_in[6];
  const float* bv = (const float*)d_in[7];
  float* out = (float*)d_out;

  char* p = (char*)d_ws;
  auto carve = [&](size_t bytes) -> char* {
    char* r = p;
    p += (bytes + 255) & ~(size_t)255;
    return r;
  };
  bf16_t* xf   = (bf16_t*)carve((size_t)NT_ * DIN * 2);
  bf16_t* mf   = (bf16_t*)carve((size_t)NT_ * DIN * 2);
  bf16_t* Wkt  = (bf16_t*)carve((size_t)H_ * DIN * DOUT * 2);
  bf16_t* Wqt  = (bf16_t*)carve((size_t)H_ * DIN * DOUT * 2);
  bf16_t* Wvt  = (bf16_t*)carve((size_t)H_ * DIN * DOUT * 2);
  bf16_t* Kf   = (bf16_t*)carve((size_t)N_ * H_ * T_ * DOUT * 2);
  bf16_t* Qf   = (bf16_t*)carve((size_t)N_ * H_ * T_ * DOUT * 2);
  bf16_t* Vt   = (bf16_t*)carve((size_t)N_ * H_ * T_ * DOUT * 2);
  float*  att  = (float*) carve((size_t)N_ * H_ * T_ * T_ * 4);
  bf16_t* attb = (bf16_t*)carve((size_t)N_ * H_ * T_ * T_ * 2);

  // 1) pack activations to bf16 [NT, DIN]
  pack_nctv<<<(N_ * T_ * C_IN * V_) / 256, 256, 0, stream>>>(x, xf);
  pack_nctv<<<(N_ * T_ * C_IN * V_) / 256, 256, 0, stream>>>(mm, mf);

  // 2) transpose + convert weights to bf16 [h][o][f]
  dim3 tg(DIN / 32, DOUT / 32, H_);
  transpose_w<<<tg, dim3(32, 8), 0, stream>>>(Wk, Wkt);
  transpose_w<<<tg, dim3(32, 8), 0, stream>>>(Wq, Wqt);
  transpose_w<<<tg, dim3(32, 8), 0, stream>>>(Wv, Wvt);

  // 3) per-head projections (WMMA): keys/queries row-major, values transposed
  dim3 pg(NT_ / (MI_ * 16), DOUT / (NJ_ * 16), H_);
  proj_gemm<0><<<pg, 32, 0, stream>>>(xf, Wkt, bk, Kf);
  proj_gemm<0><<<pg, 32, 0, stream>>>(mf, Wqt, bq, Qf);
  proj_gemm<1><<<pg, 32, 0, stream>>>(mf, Wvt, bv, Vt);

  // 4) attention scores (WMMA)
  attn_scores<<<dim3(T_ / (MI_ * 16), T_ / (NJ_ * 16), N_ * H_), 32, 0, stream>>>(Qf, Kf, att);

  // 5) softmax over heads
  softmax_heads<<<(N_ * T_ * T_) / 256, 256, 0, stream>>>(att, attb);

  // 6) apply attention (WMMA) + final scatter
  attn_apply<<<dim3(T_ / (MI_ * 16), DOUT / (NJ_ * 16), N_ * H_), 32, 0, stream>>>(attb, Vt, out);
}
